// attention_block_39127152066641
// MI455X (gfx1250) — matrix operand order repaired and checked by reading, not compiled, not run on hardware
//
#include <hip/hip_runtime.h>
#include <hip/hip_fp16.h>

typedef __attribute__((ext_vector_type(16))) _Float16 v16h;
typedef __attribute__((ext_vector_type(8)))  float    v8f;

constexpr int Bn = 4, Tn = 2048, Cn = 512, Hn = 8, HSn = 64;
constexpr float LN_EPS = 1e-5f;
constexpr float SCALE  = 0.044194173824159216f; // C^-0.5 = 1/sqrt(512)

union F16x16 { v16h v; uint4 q[2]; _Float16 h[16]; };
union F32x8  { v8f  v; float f[8]; };

// ---------------------------------------------------------------------------
// Fragment helpers (CDNA5 wave32 WMMA layouts, 16x16x32 f16 -> f32)
//
// A (16x32, MxK): lane = m + 16*hb ; elements 0..7 -> K = hb*8 + e,
//                 elements 8..15 -> K = 16 + hb*8 + (e-8)
// B (32x16, KxN): lane = n + 16*hb ; same K order as A
// C/D (16x16):    lane = n + 16*hb ; vgpr r -> row m = r + 8*hb, col n
// ---------------------------------------------------------------------------
__device__ inline v8f wmma16(v16h a, v16h b, v8f c) {
  return __builtin_amdgcn_wmma_f32_16x16x32_f16(false, a, false, b,
                                                (short)0, c, false, false);
}

// A-fragment from row-major f16 [M x >=K] matrix (ld in halfs, k0 mult of 32)
__device__ inline v16h load_a_rm(const _Float16* __restrict__ base, int ld, int k0) {
  const int lane = threadIdx.x & 31;
  const int m = lane & 15, hb = lane >> 4;
  const _Float16* p = base + (size_t)m * ld + k0 + hb * 8;
  F16x16 f;
  f.q[0] = *(const uint4*)(p);        // K = k0 + hb*8 + [0..7]
  f.q[1] = *(const uint4*)(p + 16);   // K = k0 + 16 + hb*8 + [0..7]
  return f.v;
}

// B-fragment where B[k][n] is stored with k contiguous per n-row:
// base + n*ld + k. (K row-major for K^T, V transposed for P*V)
__device__ inline v16h load_b_rows(const _Float16* __restrict__ base, int ld, int k0) {
  const int lane = threadIdx.x & 31;
  const int nn = lane & 15, hb = lane >> 4;
  const _Float16* p = base + (size_t)nn * ld + k0 + hb * 8;
  F16x16 f;
  f.q[0] = *(const uint4*)(p);        // K = k0 + hb*8 + [0..7]
  f.q[1] = *(const uint4*)(p + 16);   // K = k0 + 16 + hb*8 + [0..7]
  return f.v;
}

// B-fragment from pre-packed weights: 32B contiguous per lane per (kt,nt) block
__device__ inline v16h load_b_packed(const _Float16* __restrict__ W,
                                     int kt, int nt, int ntiles) {
  const int lane = threadIdx.x & 31;
  const _Float16* p = W + (((size_t)(kt * ntiles + nt) * 32 + lane) << 4);
  F16x16 f;
  f.q[0] = *(const uint4*)(p);
  f.q[1] = *(const uint4*)(p + 8);
  return f.v;
}

// Map a column index kl (0..31) to A-frag (half-wave hb, element e)
__device__ inline void afrag_pos(int kl, int& hb, int& e) {
  hb = (kl >> 3) & 1;
  e  = (kl & 7) + ((kl >> 4) << 3);
}

// ---------------------------------------------------------------------------
// K0: pack fp32 [K x N] weight into f16 B-fragment order
// ---------------------------------------------------------------------------
__global__ __launch_bounds__(256) void pack_b(const float* __restrict__ W,
                                              _Float16* __restrict__ o,
                                              int K, int N) {
  const int i = blockIdx.x * 256 + threadIdx.x;
  if (i >= K * N) return;
  const int e = i & 15, lane = (i >> 4) & 31, blk = i >> 9;
  const int ntiles = N >> 4;
  const int kt = blk / ntiles, nt = blk - kt * ntiles;
  const int k = kt * 32 + ((lane >> 4) << 3) + (e & 7) + ((e >> 3) << 4);
  const int n = nt * 16 + (lane & 15);
  o[i] = (_Float16)W[(size_t)k * N + n];
}

// ---------------------------------------------------------------------------
// K1/K4: fused LayerNorm, fp32 in -> f16 out. One wave per 512-wide row.
// ---------------------------------------------------------------------------
__global__ __launch_bounds__(256) void ln_rows(const float* __restrict__ x,
                                               const float* __restrict__ g,
                                               const float* __restrict__ b,
                                               _Float16* __restrict__ o) {
  const int row = blockIdx.x * blockDim.y + threadIdx.y;
  const int lane = threadIdx.x;
  const float* xr = x + (size_t)row * Cn;
  float v[16];
  float s = 0.f;
#pragma unroll
  for (int i = 0; i < 16; ++i) { v[i] = xr[lane + i * 32]; s += v[i]; }
#pragma unroll
  for (int o2 = 1; o2 < 32; o2 <<= 1) s += __shfl_xor(s, o2, 32);
  const float mu = s * (1.f / Cn);
  float vs = 0.f;
#pragma unroll
  for (int i = 0; i < 16; ++i) { const float d = v[i] - mu; vs += d * d; }
#pragma unroll
  for (int o2 = 1; o2 < 32; o2 <<= 1) vs += __shfl_xor(vs, o2, 32);
  const float rs = rsqrtf(vs * (1.f / Cn) + LN_EPS);
  _Float16* orow = o + (size_t)row * Cn;
#pragma unroll
  for (int i = 0; i < 16; ++i) {
    const int ci = lane + i * 32;
    orow[ci] = (_Float16)((v[i] - mu) * rs * g[ci] + b[ci]);
  }
}

// ---------------------------------------------------------------------------
// K2: Q/K/V projection. One wave per (16-row tile, head, {q,k,v}, batch).
// 16x64 output tile, K=512 -> 64 WMMAs. V is stored transposed [HS x T].
// ---------------------------------------------------------------------------
__global__ __launch_bounds__(32) void qkv_gemm(const _Float16* __restrict__ h16,
                                               const _Float16* __restrict__ wpk,
                                               _Float16* __restrict__ q16,
                                               _Float16* __restrict__ k16,
                                               _Float16* __restrict__ vt16) {
  const int m0  = blockIdx.x * 16;
  const int typ = blockIdx.y / Hn;     // 0=q 1=k 2=v
  const int h   = blockIdx.y % Hn;
  const int b   = blockIdx.z;
  const int lane = threadIdx.x;
  const int hi = lane >> 4, n = lane & 15;
  const _Float16* A = h16 + ((size_t)b * Tn + m0) * Cn;
  const _Float16* W = wpk + ((size_t)typ * Hn + h) * Cn * HSn;
  F32x8 c[4] = {};
  for (int kt = 0; kt < Cn / 32; ++kt) {
    const v16h a = load_a_rm(A, Cn, kt * 32);
#pragma unroll
    for (int nt = 0; nt < 4; ++nt)
      c[nt].v = wmma16(a, load_b_packed(W, kt, nt, 4), c[nt].v);
  }
  if (typ < 2) {
    _Float16* O = (typ == 0 ? q16 : k16) + (size_t)(b * Hn + h) * Tn * HSn;
#pragma unroll
    for (int nt = 0; nt < 4; ++nt)
#pragma unroll
      for (int r = 0; r < 8; ++r)
        O[(size_t)(m0 + r + 8 * hi) * HSn + nt * 16 + n] = (_Float16)c[nt].f[r];
  } else {
    _Float16* O = vt16 + (size_t)(b * Hn + h) * HSn * Tn;   // [d][t]
#pragma unroll
    for (int nt = 0; nt < 4; ++nt)
#pragma unroll
      for (int r = 0; r < 8; ++r)
        O[(size_t)(nt * 16 + n) * Tn + (m0 + r + 8 * hi)] = (_Float16)c[nt].f[r];
  }
}

// ---------------------------------------------------------------------------
// K3: causal flash attention + residual. One wave per (16 query rows, h, b).
// S-tiles of 32 keys; online softmax in fp32; P routed C-frag -> A-frag via
// a 1KB LDS tile stored directly in A-frag-packed order. Writes x + attn.
// ---------------------------------------------------------------------------
__global__ __launch_bounds__(32) void flash_attn(const _Float16* __restrict__ qg,
                                                 const _Float16* __restrict__ kg,
                                                 const _Float16* __restrict__ vtg,
                                                 const float* __restrict__ x,
                                                 float* __restrict__ out) {
  const int m0 = blockIdx.x * 16;
  const int h = blockIdx.y, b = blockIdx.z;
  const int lane = threadIdx.x;
  const int hi = lane >> 4, n = lane & 15;
  const _Float16* Q  = qg  + (size_t)(b * Hn + h) * Tn * HSn;
  const _Float16* Kp = kg  + (size_t)(b * Hn + h) * Tn * HSn;
  const _Float16* Vt = vtg + (size_t)(b * Hn + h) * HSn * Tn;

  const v16h qa0 = load_a_rm(Q + (size_t)m0 * HSn, HSn, 0);
  const v16h qa1 = load_a_rm(Q + (size_t)m0 * HSn, HSn, 32);

  F32x8 acc[4] = {};
  float mrow[8], lrow[8];
  const float NEG_INF = -__builtin_inff();
#pragma unroll
  for (int r = 0; r < 8; ++r) { mrow[r] = NEG_INF; lrow[r] = 0.f; }

  __shared__ __align__(32) _Float16 pl[512];  // 16x32 P tile, A-frag packed

  const int send = m0 + 16;                    // causal bound (exclusive)
  for (int s0 = 0; s0 < send; s0 += 32) {
    // scores: two 16x16 C-frags over HS=64 (2 WMMAs each)
    F32x8 sf[2];
#pragma unroll
    for (int sub = 0; sub < 2; ++sub) {
      v8f cz = {};
      cz = wmma16(qa0, load_b_rows(Kp + (size_t)(s0 + sub * 16) * HSn, HSn, 0),  cz);
      cz = wmma16(qa1, load_b_rows(Kp + (size_t)(s0 + sub * 16) * HSn, HSn, 32), cz);
      sf[sub].v = cz;
    }
    // online softmax per row (rows 0..7 lanes 0..15, rows 8..15 lanes 16..31)
#pragma unroll
    for (int r = 0; r < 8; ++r) {
      const int row = m0 + r + 8 * hi;
      float v0 = (s0 + n      <= row) ? sf[0].f[r] * SCALE : NEG_INF;
      float v1 = (s0 + 16 + n <= row) ? sf[1].f[r] * SCALE : NEG_INF;
      float mx = fmaxf(v0, v1);
#pragma unroll
      for (int o2 = 1; o2 < 16; o2 <<= 1) mx = fmaxf(mx, __shfl_xor(mx, o2, 32));
      const float mn = fmaxf(mrow[r], mx);
      const float p0 = __expf(v0 - mn), p1 = __expf(v1 - mn);
      float ls = p0 + p1;
#pragma unroll
      for (int o2 = 1; o2 < 16; o2 <<= 1) ls += __shfl_xor(ls, o2, 32);
      const float alpha = __expf(mrow[r] - mn);   // 0 on first tile
      lrow[r] = lrow[r] * alpha + ls;
      mrow[r] = mn;
#pragma unroll
      for (int nt = 0; nt < 4; ++nt) acc[nt].f[r] *= alpha;
      // scatter P into LDS in A-frag-packed order
      const int m = r + 8 * hi;
      int hb, e;
      afrag_pos(n, hb, e);
      pl[(((hb << 4) + m) << 4) + e] = (_Float16)p0;
      afrag_pos(16 + n, hb, e);
      pl[(((hb << 4) + m) << 4) + e] = (_Float16)p1;
    }
    __syncthreads();
    const v16h pa = *(const v16h*)(pl + (lane << 4));   // 2x ds_read_b128
#pragma unroll
    for (int nt = 0; nt < 4; ++nt)
      acc[nt].v = wmma16(pa, load_b_rows(Vt + (size_t)(nt * 16) * Tn, Tn, s0),
                         acc[nt].v);
    __syncthreads();
  }
  // epilogue: out = x + softmax(S) V   (concat heads)
#pragma unroll
  for (int nt = 0; nt < 4; ++nt)
#pragma unroll
    for (int r = 0; r < 8; ++r) {
      const int t = m0 + r + 8 * hi;
      const size_t idx = ((size_t)b * Tn + t) * Cn + h * HSn + nt * 16 + n;
      out[idx] = x[idx] + acc[nt].f[r] / lrow[r];
    }
}

// ---------------------------------------------------------------------------
// K5: fused FFN. One wave per 16-row tile: act = relu(h2@w1+b1) staged in
// 64KB LDS (A-frag packed), then out += act@w2 + b2. 4096 WMMAs per wave.
// ---------------------------------------------------------------------------
__global__ __launch_bounds__(32) void ffn_fused(const _Float16* __restrict__ h2,
                                                const _Float16* __restrict__ w1p,
                                                const _Float16* __restrict__ w2p,
                                                const float* __restrict__ b1,
                                                const float* __restrict__ b2,
                                                float* __restrict__ out) {
  const int m0 = blockIdx.x * 16;
  const int lane = threadIdx.x;
  const int hi = lane >> 4, n = lane & 15;
  __shared__ __align__(32) _Float16 act[16 * 2048];   // 64 KB, A-frag packed
  const _Float16* A = h2 + (size_t)m0 * Cn;
  v16h a_all[16];
#pragma unroll
  for (int kt = 0; kt < 16; ++kt) a_all[kt] = load_a_rm(A, Cn, kt * 32);

  // GEMM1: [16x512] x [512x2048], bias + relu, into LDS
  for (int nb = 0; nb < 128; ++nb) {
    F32x8 c = {};
#pragma unroll
    for (int kt = 0; kt < 16; ++kt)
      c.v = wmma16(a_all[kt], load_b_packed(w1p, kt, nb, 128), c.v);
    const float bias = b1[nb * 16 + n];
    const int kcol = nb * 16 + n;
    const int ks = kcol >> 5;
    int hb, e;
    afrag_pos(kcol & 31, hb, e);
#pragma unroll
    for (int r = 0; r < 8; ++r) {
      float v = c.f[r] + bias;
      v = v > 0.f ? v : 0.f;
      act[((ks * 32 + (hb << 4) + (r + 8 * hi)) << 4) + e] = (_Float16)v;
    }
  }
  __syncthreads();

  // GEMM2: [16x2048] x [2048x512], bias, accumulate into out (holds x2)
  for (int ntb = 0; ntb < 8; ++ntb) {
    F32x8 c[4] = {};
    for (int kt = 0; kt < 64; ++kt) {
      const v16h a = *(const v16h*)(act + ((kt * 32 + lane) << 4));
#pragma unroll
      for (int j = 0; j < 4; ++j)
        c[j].v = wmma16(a, load_b_packed(w2p, kt, ntb * 4 + j, 32), c[j].v);
    }
#pragma unroll
    for (int j = 0; j < 4; ++j) {
      const int col = (ntb * 4 + j) * 16 + n;
      const float bias = b2[col];
#pragma unroll
      for (int r = 0; r < 8; ++r) {
        const size_t idx = (size_t)(m0 + r + 8 * hi) * Cn + col;
        out[idx] += c[j].f[r] + bias;
      }
    }
  }
}

// ---------------------------------------------------------------------------
extern "C" void kernel_launch(void* const* d_in, const int* in_sizes, int n_in,
                              void* d_out, int out_size, void* d_ws, size_t ws_size,
                              hipStream_t stream) {
  const float* x   = (const float*)d_in[0];
  const float* wq  = (const float*)d_in[1];
  const float* wk  = (const float*)d_in[2];
  const float* wv  = (const float*)d_in[3];
  const float* w1  = (const float*)d_in[4];
  const float* b1  = (const float*)d_in[5];
  const float* w2  = (const float*)d_in[6];
  const float* b2  = (const float*)d_in[7];
  const float* g1  = (const float*)d_in[8];
  const float* be1 = (const float*)d_in[9];
  const float* g2  = (const float*)d_in[10];
  const float* be2 = (const float*)d_in[11];
  float* out = (float*)d_out;

  char* ws = (char*)d_ws;
  size_t off = 0;
  auto carve = [&](size_t bytes) -> char* {
    char* p = ws + off;
    off += (bytes + 255) & ~(size_t)255;
    return p;
  };
  _Float16* h16   = (_Float16*)carve((size_t)Bn * Tn * Cn * 2);        // LN1 out
  _Float16* q16   = (_Float16*)carve((size_t)Bn * Hn * Tn * HSn * 2);  // Q [t][d]
  _Float16* k16   = (_Float16*)carve((size_t)Bn * Hn * Tn * HSn * 2);  // K [t][d]
  _Float16* vt16  = (_Float16*)carve((size_t)Bn * Hn * HSn * Tn * 2);  // V^T [d][t]
  _Float16* h2    = (_Float16*)carve((size_t)Bn * Tn * Cn * 2);        // LN2 out
  _Float16* wqkvp = (_Float16*)carve((size_t)3 * Hn * Cn * HSn * 2);   // packed
  _Float16* w1p   = (_Float16*)carve((size_t)Cn * 4 * Cn * 2);         // packed
  _Float16* w2p   = (_Float16*)carve((size_t)4 * Cn * Cn * 2);         // packed

  // K0: pack weights into B-fragment layout (f16)
  const float* wsrc[3] = {wq, wk, wv};
  for (int t = 0; t < 3; ++t)
    for (int h = 0; h < Hn; ++h)
      pack_b<<<(Cn * HSn + 255) / 256, 256, 0, stream>>>(
          wsrc[t] + (size_t)h * Cn * HSn,
          wqkvp + ((size_t)t * Hn + h) * Cn * HSn, Cn, HSn);
  pack_b<<<(Cn * 4 * Cn + 255) / 256, 256, 0, stream>>>(w1, w1p, Cn, 4 * Cn);
  pack_b<<<(4 * Cn * Cn + 255) / 256, 256, 0, stream>>>(w2, w2p, 4 * Cn, Cn);

  // K1: LN1
  ln_rows<<<dim3(Bn * Tn / 8), dim3(32, 8), 0, stream>>>(x, g1, be1, h16);
  // K2: QKV projections
  qkv_gemm<<<dim3(Tn / 16, 3 * Hn, Bn), 32, 0, stream>>>(h16, wqkvp, q16, k16, vt16);
  // K3: causal flash attention + residual -> out holds x2
  flash_attn<<<dim3(Tn / 16, Hn, Bn), 32, 0, stream>>>(q16, k16, vt16, x, out);
  // K4: LN2 on x2
  ln_rows<<<dim3(Bn * Tn / 8), dim3(32, 8), 0, stream>>>(out, g2, be2, h2);
  // K5: fused FFN, accumulates into out -> x2 + ff
  ffn_fused<<<dim3(Bn * Tn / 16), 32, 0, stream>>>(h2, w1p, w2p, b1, b2, out);
}
